// Transformer_21492016349890
// MI455X (gfx1250) — matrix operand order repaired and checked by reading, not compiled, not run on hardware
//
#include <hip/hip_runtime.h>

typedef __attribute__((ext_vector_type(16))) _Float16 v16h;
typedef __attribute__((ext_vector_type(8)))  _Float16 v8h;
typedef __attribute__((ext_vector_type(8)))  float    v8f;

union Frag16 { v16h v; v8h h[2]; };

#define T_    56
#define E_    270
#define H_    6
#define HS_   45
#define L_    6
#define EP    288              // E padded to multiple of 32
#define TPAD  64               // T padded to multiple of 16
#define WMAT  (EP * EP)        // 82944 halves / matrix / layer
#define WLAY  ((size_t)L_ * WMAT)

// ---------------------------------------------------------------------------
// Fragment layouts (ISA 7.12.2), f16 16x16x32, wave32:
// A (16x32): lanes 0-15: halves 0-7 -> K 0-7,  halves 8-15 -> K 16-23
//            lanes16-31: halves 0-7 -> K 8-15, halves 8-15 -> K 24-31
// B (32x16): lane n&15 holds 16 K of column n, same K order as A;
//            lanes 0-15 K 0-7,16-23, lanes 16-31 K 8-15,24-31 -> read from [n][k].
// ---------------------------------------------------------------------------

// One 16x16 tile, accumulate over KT k-tiles (used for small LDS-B GEMMs).
template <int KT>
static __device__ __forceinline__ v8f gemm_tile(const _Float16* __restrict__ A, int aStride, int mBase,
                                                const _Float16* __restrict__ B, int bStride, int nBase,
                                                int lane)
{
    const int nl = lane & 15;
    const int hi = (lane >> 4) & 1;
    const _Float16* arow = A + (mBase + nl) * aStride;
    const _Float16* brow = B + (nBase + nl) * bStride;
    v8f c = {};
#pragma unroll
    for (int kt = 0; kt < KT; ++kt) {
        Frag16 a, b;
        const int ka = kt * 32 + (hi ? 8 : 0);
        a.h[0] = *(const v8h*)(arow + ka);
        a.h[1] = *(const v8h*)(arow + ka + 16);
        const int kb = kt * 32 + (hi ? 8 : 0);
        b.h[0] = *(const v8h*)(brow + kb);
        b.h[1] = *(const v8h*)(brow + kb + 16);
        c = __builtin_amdgcn_wmma_f32_16x16x32_f16(false, a.v, false, b.v,
                                                   (short)0, c, false, false);
    }
    return c;
}

// Column job: load B fragments (all KT k-tiles) ONCE into registers, then
// sweep MT m-tiles reusing them; A fragments re-read from LDS (cheap ds ops).
// Cuts L2 weight traffic by MT x vs tile-per-wave.
template <int KT, int MT>
static __device__ __forceinline__ void gemm_col(const _Float16* __restrict__ A, int aStride, int mBase,
                                                const _Float16* __restrict__ B, int bStride, int nBase,
                                                int lane, v8f c[MT])
{
    const int nl = lane & 15;
    const int hi = (lane >> 4) & 1;
    const _Float16* brow = B + (nBase + nl) * bStride;
    Frag16 bf[KT];
#pragma unroll
    for (int kt = 0; kt < KT; ++kt) {
        const int kb = kt * 32 + (hi ? 8 : 0);
        bf[kt].h[0] = *(const v8h*)(brow + kb);
        bf[kt].h[1] = *(const v8h*)(brow + kb + 16);
    }
#pragma unroll
    for (int mt = 0; mt < MT; ++mt) {
        const _Float16* arow = A + (mBase + mt * 16 + nl) * aStride;
        v8f acc = {};
#pragma unroll
        for (int kt = 0; kt < KT; ++kt) {
            Frag16 a;
            const int ka = kt * 32 + (hi ? 8 : 0);
            a.h[0] = *(const v8h*)(arow + ka);
            a.h[1] = *(const v8h*)(arow + ka + 16);
            acc = __builtin_amdgcn_wmma_f32_16x16x32_f16(false, a.v, false, bf[kt].v,
                                                         (short)0, acc, false, false);
        }
        c[mt] = acc;
    }
}

// fp32 [L][270][270] -> f16 [L][288][288], zero padded ([out][in] kept as-is).
__global__ void pack_weights(const float* __restrict__ src, _Float16* __restrict__ dst, int total)
{
    int i = blockIdx.x * 256 + threadIdx.x;
    if (i >= total) return;
    int l = i / WMAT;
    int r = (i / EP) % EP;
    int c = i % EP;
    float v = 0.f;
    if (r < E_ && c < E_) v = src[(size_t)l * E_ * E_ + r * E_ + c];
    dst[i] = (_Float16)v;
}

__global__ __launch_bounds__(256) void txf_kernel(
    const int*   __restrict__ xb,  const float* __restrict__ emb,
    const float* __restrict__ b1,  const float* __restrict__ b2,
    const float* __restrict__ Wf,  const float* __restrict__ bf,
    const _Float16* __restrict__ WqH, const _Float16* __restrict__ WkH,
    const _Float16* __restrict__ WvH, const _Float16* __restrict__ W1H,
    const _Float16* __restrict__ W2H, float* __restrict__ out)
{
    extern __shared__ __align__(16) unsigned char smem_raw[];
    _Float16* sX  = (_Float16*)smem_raw;        // [64][288]  activations / concat-heads
    _Float16* sQ  = sX + TPAD * EP;             // [6][64(t)][64(d)]   q  (d>=45 pad zero)
    _Float16* sK  = sQ + H_ * 64 * 64;          // [6][64(s)][64(d)]   k
    _Float16* sV  = sK + H_ * 64 * 64;          // [6][64(d)][64(s)]   v transposed
    _Float16* sA  = sV + H_ * 64 * 64;          // [64(t)][64(s)]      attn (pads zero)
    float*    sS  = (float*)(sA + 64 * 64);     // [64][64] f32 scores / reduce buf
    _Float16* sHd = (_Float16*)(sS + 64 * 64);  // [64][288] FF hidden

    const int tid  = threadIdx.x;
    const int lane = tid & 31;
    const int wv   = tid >> 5;
    const int nl   = lane & 15;
    const int hi   = (lane >> 4) & 1;
    const int b    = blockIdx.x;
    const int* ids = xb + (size_t)b * T_;
    const float scale = 0.1490711984999860f;    // 45^-0.5

    // zero q/k/v/attn once: pads stay zero for all 6 layers (valid regions
    // are fully rewritten each layer, pads never touched again)
    for (int i = tid; i < H_ * 64 * 64 * 3 + 64 * 64; i += 256) sQ[i] = (_Float16)0.f;

    // embedding gather -> f16, zero-padded to [64][288]
    for (int i = tid; i < TPAD * EP; i += 256) {
        int t = i / EP, e = i - t * EP;
        float v = 0.f;
        if (t < T_ && e < E_) v = emb[(size_t)ids[t] * E_ + e];
        sX[i] = (_Float16)v;
    }
    __syncthreads();

    for (int l = 0; l < L_; ++l) {
        const size_t wOff = (size_t)l * WMAT;

        // ---- QKV: 54 column jobs (3 matrices x 18 N-tiles), B regs reused 4x
        for (int job = wv; job < 54; job += 8) {
            int mtx = job / 18, nt = job % 18;
            const _Float16* W = (mtx == 0 ? WqH : (mtx == 1 ? WkH : WvH)) + wOff;
            // prefetch next job's weight panel rows (per-lane row start)
            int nj = job + 8;
            if (nj < 54) {
                int nmtx = nj / 18, nnt = nj % 18;
                const _Float16* Wn = (nmtx == 0 ? WqH : (nmtx == 1 ? WkH : WvH)) + wOff;
                __builtin_prefetch(Wn + (nnt * 16 + nl) * EP, 0, 0);
            }
            v8f c[4];
            gemm_col<9, 4>(sX, EP, 0, W, EP, nt * 16, lane, c);
            int n = nt * 16 + nl;
            if (n < E_) {
                int h = n / HS_, d = n % HS_;
#pragma unroll
                for (int mt = 0; mt < 4; ++mt) {
                    int mo = mt * 16 + hi * 8;
#pragma unroll
                    for (int r = 0; r < 8; ++r) {
                        int m = mo + r;
                        if (m < T_) {
                            if (mtx == 0)      sQ[h * 4096 + m * 64 + d] = (_Float16)c[mt][r];
                            else if (mtx == 1) sK[h * 4096 + m * 64 + d] = (_Float16)c[mt][r];
                            else               sV[h * 4096 + d * 64 + m] = (_Float16)c[mt][r]; // transposed
                        }
                    }
                }
            }
        }
        __syncthreads();

        // ---- attention, one head at a time (B operands in LDS: tile-per-wave)
        for (int hd = 0; hd < H_; ++hd) {
            // scores[t][s] = q . k^T   (K = HS padded to 64)
            for (int tile = wv; tile < 16; tile += 8) {
                int mt = tile >> 2, nt = tile & 3;
                v8f c = gemm_tile<2>(sQ + hd * 4096, 64, mt * 16,
                                     sK + hd * 4096, 64, nt * 16, lane);
                int n = nt * 16 + nl, mo = mt * 16 + hi * 8;
                if (n < T_) {
#pragma unroll
                    for (int r = 0; r < 8; ++r) {
                        int m = mo + r;
                        if (m < T_) sS[m * 64 + n] = c[r];
                    }
                }
            }
            __syncthreads();

            // softmax over the QUERY axis t (reference: softmax(dim=1) -> axis=2)
            if (tid < T_) {
                int s = tid;
                float mx = -3.4e38f;
                for (int t = 0; t < T_; ++t) mx = fmaxf(mx, sS[t * 64 + s] * scale);
                float sum = 0.f;
                for (int t = 0; t < T_; ++t) sum += __expf(sS[t * 64 + s] * scale - mx);
                float inv = 1.f / sum;
                for (int t = 0; t < T_; ++t)
                    sA[t * 64 + s] = (_Float16)(__expf(sS[t * 64 + s] * scale - mx) * inv);
            }
            __syncthreads();

            // val[t][d] = attn x v ; write concat-heads into sX (x fully consumed)
            for (int tile = wv; tile < 12; tile += 8) {
                int mt = tile / 3, nt = tile % 3;
                v8f c = gemm_tile<2>(sA, 64, mt * 16,
                                     sV + hd * 4096, 64, nt * 16, lane);
                int d = nt * 16 + nl, mo = mt * 16 + hi * 8;
                if (d < HS_) {
#pragma unroll
                    for (int r = 0; r < 8; ++r) {
                        int m = mo + r;
                        if (m < T_) sX[m * EP + hd * HS_ + d] = (_Float16)c[r];
                    }
                }
            }
            __syncthreads();
        }

        // ---- FF1: hidden = relu(comb x W1^T + b1) -> sHd (full [64][288] incl. pads)
        //      36 jobs = 18 N-tiles x 2 M-halves, B regs reused 2x
        for (int job = wv; job < 36; job += 8) {
            int mh = job / 18, nt = job % 18;
            int nj = job + 8;
            if (nj < 36) __builtin_prefetch(W1H + wOff + ((nj % 18) * 16 + nl) * EP, 0, 0);
            v8f c[2];
            gemm_col<9, 2>(sX, EP, mh * 32, W1H + wOff, EP, nt * 16, lane, c);
            int n = nt * 16 + nl;
            float bias = (n < E_) ? b1[l * E_ + n] : 0.f;
#pragma unroll
            for (int mt = 0; mt < 2; ++mt) {
                int mo = mh * 32 + mt * 16 + hi * 8;
#pragma unroll
                for (int r = 0; r < 8; ++r) {
                    int m = mo + r;
                    float v = (m < T_ && n < E_) ? fmaxf(c[mt][r] + bias, 0.f) : 0.f;
                    sHd[m * EP + n] = (_Float16)v;
                }
            }
        }
        __syncthreads();

        // ---- FF2: x = hidden x W2^T + b2 -> sX (pads untouched, remain zero)
        for (int job = wv; job < 36; job += 8) {
            int mh = job / 18, nt = job % 18;
            int nj = job + 8;
            if (nj < 36) __builtin_prefetch(W2H + wOff + ((nj % 18) * 16 + nl) * EP, 0, 0);
            v8f c[2];
            gemm_col<9, 2>(sHd, EP, mh * 32, W2H + wOff, EP, nt * 16, lane, c);
            int n = nt * 16 + nl;
            if (n < E_) {
                float bias = b2[l * E_ + n];
#pragma unroll
                for (int mt = 0; mt < 2; ++mt) {
                    int mo = mh * 32 + mt * 16 + hi * 8;
#pragma unroll
                    for (int r = 0; r < 8; ++r) {
                        int m = mo + r;
                        if (m < T_) sX[m * EP + n] = (_Float16)(c[mt][r] + bias);
                    }
                }
            }
        }
        __syncthreads();
    }

    // ---- classifier: out[b] = flat(x) . Wf + bf ----
    float acc = 0.f;
    for (int i = tid; i < T_ * E_; i += 256) {
        int t = i / E_, e = i - t * E_;
        acc += (float)sX[t * EP + e] * Wf[i];
    }
    sS[tid] = acc;
    __syncthreads();
    for (int s = 128; s > 0; s >>= 1) {
        if (tid < s) sS[tid] += sS[tid + s];
        __syncthreads();
    }
    if (tid == 0) out[b] = sS[0] + bf[0];
}

extern "C" void kernel_launch(void* const* d_in, const int* in_sizes, int n_in,
                              void* d_out, int out_size, void* d_ws, size_t ws_size,
                              hipStream_t stream)
{
    (void)n_in; (void)out_size; (void)ws_size;
    const int*   xb  = (const int*)  d_in[0];
    const float* emb = (const float*)d_in[2];
    const float* Wq  = (const float*)d_in[3];
    const float* Wk  = (const float*)d_in[4];
    const float* Wv  = (const float*)d_in[5];
    const float* W1  = (const float*)d_in[6];
    const float* b1  = (const float*)d_in[7];
    const float* W2  = (const float*)d_in[8];
    const float* b2  = (const float*)d_in[9];
    const float* Wf  = (const float*)d_in[10];
    const float* bf  = (const float*)d_in[11];
    float* out = (float*)d_out;

    _Float16* WqH = (_Float16*)d_ws;
    _Float16* WkH = WqH + WLAY;
    _Float16* WvH = WkH + WLAY;
    _Float16* W1H = WvH + WLAY;
    _Float16* W2H = W1H + WLAY;

    const int total = (int)WLAY;               // 497664 halves per tensor
    const int pb = (total + 255) / 256;
    pack_weights<<<pb, 256, 0, stream>>>(Wq, WqH, total);
    pack_weights<<<pb, 256, 0, stream>>>(Wk, WkH, total);
    pack_weights<<<pb, 256, 0, stream>>>(Wv, WvH, total);
    pack_weights<<<pb, 256, 0, stream>>>(W1, W1H, total);
    pack_weights<<<pb, 256, 0, stream>>>(W2, W2H, total);

    const size_t shmem =
        (size_t)(TPAD * EP + 3 * H_ * 64 * 64 + 64 * 64) * sizeof(_Float16) // X,Q,K,V,attn
        + (size_t)64 * 64 * sizeof(float)                                   // scores
        + (size_t)TPAD * EP * sizeof(_Float16);                             // FF hidden
    (void)hipFuncSetAttribute((const void*)txf_kernel,
                              hipFuncAttributeMaxDynamicSharedMemorySize, (int)shmem);

    const int B = in_sizes[0] / T_;            // 4096
    txf_kernel<<<B, 256, shmem, stream>>>(xb, emb, b1, b2, Wf, bf,
                                          WqH, WkH, WvH, W1H, W2H, out);
}
